// Attention_87625922773493
// MI455X (gfx1250) — matrix operand order repaired and checked by reading, not compiled, not run on hardware
//
#include <hip/hip_runtime.h>

// ---------- problem constants (match reference: B=4, N=4096, D=256) ----------
#define BATCH 4
#define NSEQ  4096
#define DDIM  256

// K tile in LDS: 32 rows x (256 + 8 pad) bf16  (TDM pads 16B per 512B row)
#define KROW   264
#define KTILEB (32 * KROW * 2)
// V tile in LDS: 256 rows x (32 + 8 pad) bf16  (TDM pads 16B per 64B row)
#define VROW   40
#define VTILEB (256 * VROW * 2)

// ---------- types ----------
typedef __attribute__((ext_vector_type(16))) __bf16          v16bf;
typedef __attribute__((ext_vector_type(8)))  float           v8f;
typedef __attribute__((ext_vector_type(8)))  unsigned short  us8;
typedef __attribute__((ext_vector_type(4)))  unsigned short  us4;
typedef __attribute__((ext_vector_type(4)))  unsigned int    u32x4;
typedef __attribute__((ext_vector_type(8)))  int             i32x8;
typedef __attribute__((ext_vector_type(4)))  int             i32x4;

union Frag16 {            // 16x32 (A) or 32x16 (B) bf16 operand, 8 VGPRs/lane
    v16bf bf;
    us8   us[2];
};

// round-to-nearest-even f32 -> bf16 (bit pattern)
__device__ __host__ inline unsigned short f32_to_bf16(float f) {
    union { float f; unsigned int u; } c; c.f = f;
    unsigned int u = c.u;
    unsigned int r = (u + 0x7FFFu + ((u >> 16) & 1u)) >> 16;
    return (unsigned short)r;
}

// ---------------------------------------------------------------------------
// TDM 2D tile load: build D# groups per CDNA5 ISA and
// issue tensor_load_to_lds (clang-23 6-arg builtin). Wave-level op; tracked
// with TENSORcnt. data_size = 2 bytes (bf16). Optional LDS row padding.
// ---------------------------------------------------------------------------
__device__ static inline void tdm_load_2d(
    const void* gptr,          // global address of tile start
    unsigned int lds_off,      // LDS byte offset of tile dest
    unsigned int tile_w,       // tile width  (elements)
    unsigned int tile_h,       // tile height (rows)
    unsigned int tensor_w,     // tensor width  (elements, OOB bound)
    unsigned int tensor_h,     // tensor height (rows,     OOB bound)
    unsigned int row_stride,   // elements between tensor rows
    unsigned int pad_interval, // code: pad after 2^(k+1) DWORDs
    unsigned int pad_amount)   // code: insert (k+1) DWORDs of pad
{
    unsigned long long ga = (unsigned long long)(uintptr_t)gptr;
    u32x4 g0;
    g0[0] = 1u;                                               // count=1, user
    g0[1] = lds_off;                                          // lds_addr
    g0[2] = (unsigned int)ga;                                 // global_addr lo
    g0[3] = (unsigned int)((ga >> 32) & 0x01FFFFFFu) | (2u << 30); // hi | type=2
    i32x8 g1;
    g1[0] = (int)((1u << 16)                                  // data_size = 2B
                | (1u << 20)                                  // pad_enable
                | (pad_interval << 22) | (pad_amount << 25));
    g1[1] = (int)((tensor_w & 0xFFFFu) << 16);                // tensor_dim0 lo
    g1[2] = (int)(((tensor_w >> 16) & 0xFFFFu) | ((tensor_h & 0xFFFFu) << 16));
    g1[3] = (int)(((tensor_h >> 16) & 0xFFFFu) | ((tile_w & 0xFFFFu) << 16));
    g1[4] = (int)(tile_h & 0xFFFFu);                          // tile_dim1 (dim2=0)
    g1[5] = (int)row_stride;                                  // dim0_stride lo
    g1[6] = 0;                                                // stride hi, dim1_stride
    g1[7] = 0;
    i32x4 z4 = {0, 0, 0, 0};                                  // 2D: groups 2/3 zero
    i32x8 z8 = {0, 0, 0, 0, 0, 0, 0, 0};
    __builtin_amdgcn_tensor_load_to_lds(g0, g1, z4, z4, z8, 0);
}

// ---------------------------------------------------------------------------
// Kernel 1: f32 -> bf16 convert, 4 elements/thread
// ---------------------------------------------------------------------------
__global__ __launch_bounds__(256) void convert_f32_bf16_x4(
    const float* __restrict__ in, unsigned short* __restrict__ out, int n4)
{
    int i = blockIdx.x * 256 + threadIdx.x;
    if (i >= n4) return;
    float4 f = reinterpret_cast<const float4*>(in)[i];
    us4 o;
    o[0] = f32_to_bf16(f.x);
    o[1] = f32_to_bf16(f.y);
    o[2] = f32_to_bf16(f.z);
    o[3] = f32_to_bf16(f.w);
    reinterpret_cast<us4*>(out)[i] = o;
}

// ---------------------------------------------------------------------------
// Kernel 2: per-batch transpose x[N][D] (f32) -> Xt[D][N] (bf16), 32x32 tiles
// ---------------------------------------------------------------------------
__global__ __launch_bounds__(256) void transpose_f32_bf16(
    const float* __restrict__ X, unsigned short* __restrict__ Xt)
{
    __shared__ unsigned short tile[32][33];
    const int b  = blockIdx.z;
    const int n0 = blockIdx.x * 32;
    const int d0 = blockIdx.y * 32;
    const float*    Xb = X  + (size_t)b * NSEQ * DDIM;
    unsigned short* Tb = Xt + (size_t)b * DDIM * NSEQ;

    for (int i = threadIdx.y; i < 32; i += 8)
        tile[i][threadIdx.x] = f32_to_bf16(Xb[(size_t)(n0 + i) * DDIM + d0 + threadIdx.x]);
    __syncthreads();
    for (int i = threadIdx.y; i < 32; i += 8)
        Tb[(size_t)(d0 + i) * NSEQ + n0 + threadIdx.x] = tile[threadIdx.x][i];
}

// ---------------------------------------------------------------------------
// Kernel 3: Q = x @ W^T  (bf16 WMMA, f32 accumulate, bf16 out)
// ---------------------------------------------------------------------------
__global__ __launch_bounds__(128) void qgemm_bf16(
    const unsigned short* __restrict__ Xb,   // [B][N][D] bf16
    const unsigned short* __restrict__ Wb,   // [D][D]    bf16 (row e, col d)
    unsigned short* __restrict__ Qb)         // [B][N][D] bf16
{
    const int lane = threadIdx.x & 31;
    const int wave = threadIdx.x >> 5;
    const int l15  = lane & 15;
    const int half = lane >> 4;

    const int tilesPerBatch = (NSEQ / 16) * (DDIM / 16);        // 4096
    const int tile  = blockIdx.x * 4 + wave;
    const int b     = tile / tilesPerBatch;
    const int t     = tile % tilesPerBatch;
    const int rbase = (t / (DDIM / 16)) * 16;
    const int ebase = (t % (DDIM / 16)) * 16;

    const unsigned short* xrow = Xb + (size_t)b * NSEQ * DDIM + (size_t)(rbase + l15) * DDIM;
    const unsigned short* wrow = Wb + (size_t)(ebase + l15) * DDIM;

    v8f acc = {};
    #pragma unroll
    for (int kc = 0; kc < 8; ++kc) {
        Frag16 a, w;
        a.us[0] = *(const us8*)(xrow + kc * 32 + half * 8);
        a.us[1] = *(const us8*)(xrow + kc * 32 + 16 + half * 8);
        w.us[0] = *(const us8*)(wrow + kc * 32 + half * 8);
        w.us[1] = *(const us8*)(wrow + kc * 32 + 16 + half * 8);
        acc = __builtin_amdgcn_wmma_f32_16x16x32_bf16(
                  false, a.bf, false, w.bf, (short)0, acc, false, false);
    }

    unsigned short* qout = Qb + (size_t)b * NSEQ * DDIM;
    #pragma unroll
    for (int r = 0; r < 8; ++r)
        qout[(size_t)(rbase + r + 8 * half) * DDIM + ebase + l15] = f32_to_bf16(acc[r]);
}

// ---------------------------------------------------------------------------
// Kernel 4: flash attention, y = softmax(Q K^T) V  (V = x, no scaling)
// 4 waves / WG, each wave owns 16 query rows; 32 keys per iteration.
// K/V tiles DMA'd into double-buffered LDS by the Tensor Data Mover while
// the waves compute on the other buffer (s_wait_tensorcnt + barrier sync).
// ---------------------------------------------------------------------------
__global__ __launch_bounds__(128) void flash_attn_bf16(
    const unsigned short* __restrict__ Qb,   // [B][N][D] bf16
    const unsigned short* __restrict__ Kb,   // [B][N][D] bf16
    const unsigned short* __restrict__ Vt,   // [B][D][N] bf16 (V transposed)
    float* __restrict__ Y)                   // [B][N][D] f32
{
    const int lane = threadIdx.x & 31;
    const int wave = threadIdx.x >> 5;
    const int l15  = lane & 15;
    const int half = lane >> 4;

    const int b     = blockIdx.y;
    const int qbase = blockIdx.x * 64 + wave * 16;

    const unsigned short* Q = Qb + (size_t)b * NSEQ * DDIM;
    const unsigned short* K = Kb + (size_t)b * NSEQ * DDIM;
    const unsigned short* V = Vt + (size_t)b * DDIM * NSEQ;

    // double-buffered TDM destination tiles + per-wave P staging
    __shared__ alignas(16) unsigned short ldsK[2][32][KROW];    // 33,792 B
    __shared__ alignas(16) unsigned short ldsV[2][256][VROW];   // 40,960 B
    __shared__ alignas(16) unsigned short ldsP[4][16][32];      //  4,096 B

    const unsigned int ldsK_off = (unsigned int)(uintptr_t)&ldsK[0][0][0];
    const unsigned int ldsV_off = (unsigned int)(uintptr_t)&ldsV[0][0][0];

    // Resident Q fragments: 8 chunks of the D dimension (16x32 each)
    Frag16 qf[8];
    const unsigned short* qrow = Q + (size_t)(qbase + l15) * DDIM;
    #pragma unroll
    for (int kc = 0; kc < 8; ++kc) {
        qf[kc].us[0] = *(const us8*)(qrow + kc * 32 + half * 8);
        qf[kc].us[1] = *(const us8*)(qrow + kc * 32 + 16 + half * 8);
    }

    v8f   o[16];
    float mrow[8], lrow[8];
    #pragma unroll
    for (int t = 0; t < 16; ++t) o[t] = v8f{};
    #pragma unroll
    for (int r = 0; r < 8; ++r) { mrow[r] = -3.0e38f; lrow[r] = 0.0f; }

    // prologue: DMA first K/V tiles into buffer 0 (wave0: K, wave1: V)
    if (wave == 0)
        tdm_load_2d(K, ldsK_off, DDIM, 32, DDIM, NSEQ, DDIM, /*pi 128dw*/6, /*pa 4dw*/3);
    if (wave == 1)
        tdm_load_2d(V, ldsV_off, 32, DDIM, NSEQ, DDIM, NSEQ, /*pi 16dw*/3, /*pa 4dw*/3);

    for (int it = 0; it < NSEQ / 32; ++it) {
        const int nb  = it * 32;
        const int cur = it & 1;

        if (wave < 2) __builtin_amdgcn_s_wait_tensorcnt(0);   // tile[cur] landed
        __syncthreads();                                      // visible to all waves

        // kick DMA for the next key block into the other buffer
        if (nb + 32 < NSEQ) {
            const int nxt = cur ^ 1;
            if (wave == 0)
                tdm_load_2d(K + (size_t)(nb + 32) * DDIM, ldsK_off + nxt * KTILEB,
                            DDIM, 32, DDIM, NSEQ, DDIM, 6, 3);
            if (wave == 1)
                tdm_load_2d(V + (nb + 32), ldsV_off + nxt * VTILEB,
                            32, DDIM, NSEQ, DDIM, NSEQ, 3, 3);
        }

        const unsigned short* kt = &ldsK[cur][0][0];
        const unsigned short* vt = &ldsV[cur][0][0];

        // ---- S = Q K^T : two 16x16 tiles (32 keys), K operand from LDS ------
        v8f s[2];
        #pragma unroll
        for (int t = 0; t < 2; ++t) {
            v8f acc = {};
            const unsigned short* krow = kt + (size_t)(t * 16 + l15) * KROW;
            #pragma unroll
            for (int kc = 0; kc < 8; ++kc) {
                Frag16 kf;
                kf.us[0] = *(const us8*)(krow + kc * 32 + half * 8);
                kf.us[1] = *(const us8*)(krow + kc * 32 + 16 + half * 8);
                acc = __builtin_amdgcn_wmma_f32_16x16x32_bf16(
                          false, qf[kc].bf, false, kf.bf, (short)0, acc, false, false);
            }
            s[t] = acc;
        }

        // ---- online softmax (f32) ------------------------------------------
        float scale[8];
        #pragma unroll
        for (int r = 0; r < 8; ++r) {
            float mx = fmaxf(s[0][r], s[1][r]);
            #pragma unroll
            for (int m = 8; m >= 1; m >>= 1) mx = fmaxf(mx, __shfl_xor(mx, m, 32));
            const float mnew = fmaxf(mrow[r], mx);
            const float p0 = __expf(s[0][r] - mnew);
            const float p1 = __expf(s[1][r] - mnew);
            float rs = p0 + p1;
            #pragma unroll
            for (int m = 8; m >= 1; m >>= 1) rs += __shfl_xor(rs, m, 32);
            scale[r] = __expf(mrow[r] - mnew);
            lrow[r]  = lrow[r] * scale[r] + rs;
            mrow[r]  = mnew;
            ldsP[wave][r + 8 * half][l15]      = f32_to_bf16(p0);
            ldsP[wave][r + 8 * half][16 + l15] = f32_to_bf16(p1);
        }

        // wave-private LDS round trip (C layout -> A layout)
        asm volatile("s_wait_dscnt 0" ::: "memory");
        Frag16 pf;
        pf.us[0] = *(const us8*)&ldsP[wave][l15][half * 8];
        pf.us[1] = *(const us8*)&ldsP[wave][l15][16 + half * 8];

        // ---- O = O*scale + P V : 16 column tiles of D, V operand from LDS ---
        #pragma unroll
        for (int t = 0; t < 16; ++t) {
            v8f acc = o[t];
            #pragma unroll
            for (int r = 0; r < 8; ++r) acc[r] *= scale[r];
            const unsigned short* vrow = vt + (size_t)(t * 16 + l15) * VROW;
            Frag16 vf;
            vf.us[0] = *(const us8*)(vrow + half * 8);
            vf.us[1] = *(const us8*)(vrow + 16 + half * 8);
            o[t] = __builtin_amdgcn_wmma_f32_16x16x32_bf16(
                       false, pf.bf, false, vf.bf, (short)0, acc, false, false);
        }
        __syncthreads();   // all waves done with tile[cur] before it is reused
    }

    // ---- finalize: y = O / l -----------------------------------------------
    float* yb = Y + (size_t)b * NSEQ * DDIM;
    #pragma unroll
    for (int t = 0; t < 16; ++t) {
        #pragma unroll
        for (int r = 0; r < 8; ++r) {
            const int m = qbase + r + 8 * half;
            yb[(size_t)m * DDIM + t * 16 + l15] = o[t][r] / lrow[r];
        }
    }
}

// ---------------------------------------------------------------------------
// Host-side launcher
// ---------------------------------------------------------------------------
extern "C" void kernel_launch(void* const* d_in, const int* in_sizes, int n_in,
                              void* d_out, int out_size, void* d_ws, size_t ws_size,
                              hipStream_t stream) {
    (void)in_sizes; (void)n_in; (void)out_size; (void)ws_size;

    const float* x = (const float*)d_in[0];   // [B][N][D] f32
    const float* W = (const float*)d_in[1];   // [D][D]    f32
    float*       y = (float*)d_out;           // [B][N][D] f32

    const size_t nx = (size_t)BATCH * NSEQ * DDIM;   // 4,194,304
    const size_t nw = (size_t)DDIM * DDIM;           // 65,536

    // workspace layout (bf16 = ushort): Q | K | Vt | W  (~24.1 MB total)
    unsigned short* Qb = (unsigned short*)d_ws;
    unsigned short* Kb = Qb + nx;
    unsigned short* Vt = Kb + nx;
    unsigned short* Wb = Vt + nx;

    convert_f32_bf16_x4<<<(int)(nx / 4 / 256), 256, 0, stream>>>(x, Kb, (int)(nx / 4));
    convert_f32_bf16_x4<<<(int)(nw / 4 / 256), 256, 0, stream>>>(W, Wb, (int)(nw / 4));

    transpose_f32_bf16<<<dim3(NSEQ / 32, DDIM / 32, BATCH), dim3(32, 8), 0, stream>>>(x, Vt);

    qgemm_bf16<<<(BATCH * (NSEQ / 16) * (DDIM / 16)) / 4, 128, 0, stream>>>(Kb, Wb, Qb);

    flash_attn_bf16<<<dim3(NSEQ / 64, BATCH), 128, 0, stream>>>(Qb, Kb, Vt, y);
}
